// DeepFM_37349035606580
// MI455X (gfx1250) — matrix operand order repaired and checked by reading, not compiled, not run on hardware
//
#include <hip/hip_runtime.h>
#include <hip/hip_bf16.h>

typedef _Float16 h8   __attribute__((ext_vector_type(8)));
typedef _Float16 v16h __attribute__((ext_vector_type(16)));
typedef float    v8f  __attribute__((ext_vector_type(8)));

#define BATCH   16384
#define EMB     128
#define NFIELDS 4
#define DIN     512

// ---------------------------------------------------------------------------
// Weight transpose + fp32 -> f16 convert:  W (K x N, row-major) -> WT (N x K)
// ---------------------------------------------------------------------------
__global__ __launch_bounds__(256) void convert_wt(const float* __restrict__ W,
                                                  _Float16* __restrict__ WT,
                                                  int K, int N) {
    size_t i = (size_t)blockIdx.x * blockDim.x + threadIdx.x;
    size_t total = (size_t)K * N;
    if (i >= total) return;
    int n = (int)(i / K);
    int k = (int)(i % K);
    WT[i] = (_Float16)W[(size_t)k * N + n];
}

// ---------------------------------------------------------------------------
// Fused embedding gather (fp32 -> f16), FM per-row terms, fc linear term.
// One wave (32 lanes) per row; 8 rows per 256-thread block.
// ---------------------------------------------------------------------------
__global__ __launch_bounds__(256) void gather_fm(const float* __restrict__ x,
                                                 const float* __restrict__ fc,
                                                 const float* __restrict__ emb,
                                                 _Float16* __restrict__ embf16,
                                                 float* __restrict__ lin,
                                                 float* __restrict__ rowterm) {
    const int row  = blockIdx.x * 8 + (threadIdx.x >> 5);
    const int lane = threadIdx.x & 31;

    int cat[NFIELDS];
#pragma unroll
    for (int f = 0; f < NFIELDS; ++f) cat[f] = (int)x[(size_t)row * NFIELDS + f];

    float s1 = 0.f, s2 = 0.f;
#pragma unroll
    for (int t = 0; t < DIN / 32; ++t) {
        int j = t * 32 + lane;            // 0..511
        int f = j >> 7;                   // field
        float e = emb[(size_t)cat[f] * EMB + (j & (EMB - 1))];
        embf16[(size_t)row * DIN + j] = (_Float16)e;
        s1 += e;
        s2 += e * e;
    }
#pragma unroll
    for (int off = 16; off >= 1; off >>= 1) {
        s1 += __shfl_xor(s1, off, 32);
        s2 += __shfl_xor(s2, off, 32);
    }
    if (lane == 0) {
        rowterm[row] = s1 * s1 - s2;   // (sum)^2 - sum of squares, this row
        const int OFF[NFIELDS] = {0, 31360, 38167, 38185};
        float l = 0.f;
#pragma unroll
        for (int f = 0; f < NFIELDS; ++f) l += fc[cat[f] + OFF[f]];
        lin[row] = l;
    }
}

// ---------------------------------------------------------------------------
// Deterministic single-block reduction (fixed order every call).
// ---------------------------------------------------------------------------
__global__ __launch_bounds__(256) void reduce_sum(const float* __restrict__ in,
                                                  float* __restrict__ out, int n) {
    __shared__ float sm[256];
    float s = 0.f;
    for (int i = threadIdx.x; i < n; i += 256) s += in[i];
    sm[threadIdx.x] = s;
    __syncthreads();
    for (int off = 128; off >= 1; off >>= 1) {
        if ((int)threadIdx.x < off) sm[threadIdx.x] += sm[threadIdx.x + off];
        __syncthreads();
    }
    if (threadIdx.x == 0) out[0] = sm[0];
}

// ---------------------------------------------------------------------------
// WMMA GEMM:  C = relu(A * B + bias),  f16 in / f32 acc / f16 out
//   A  : M x K row-major (f16)
//   BT : N x K row-major (f16)  == B column-major
// Block tile 256(M) x 64(N), K-step 32, 8 waves. Each wave owns a 32x64
// sub-tile: 2 A fragments x 4 B fragments -> 8 WMMAs per K-step.
// B tile is double-buffered in LDS (stage k+1 while computing k; one barrier
// per K-step). LDS row stride 80 bytes: 16B-aligned b128, bank-conflict free.
// ---------------------------------------------------------------------------
#define TM 256
#define TN 64
#define TK 32
#define LDSB_STRIDE 40   // halves; 80 bytes

__device__ __forceinline__ v16h load_a_frag(const _Float16* ap, int kh) {
    union { v16h v; h8 h[2]; } u;
    u.h[0] = *(const h8*)(ap + kh * 8);        // K[0..8)  / K[8..16)
    u.h[1] = *(const h8*)(ap + 16 + kh * 8);   // K[16..24)/ K[24..32)
    return u.v;
}
__device__ __forceinline__ v16h load_b_frag(const _Float16* bp) {
    union { v16h v; h8 h[2]; } u;
    u.h[0] = *(const h8*)(bp);
    u.h[1] = *(const h8*)(bp + 16);
    return u.v;
}

__global__ __launch_bounds__(256) void gemm_relu_wmma(
    const _Float16* __restrict__ A, const _Float16* __restrict__ BT,
    const float* __restrict__ bias, _Float16* __restrict__ C,
    int M, int N, int K) {
    __shared__ _Float16 ldsB[2][TN * LDSB_STRIDE];

    const int tid  = threadIdx.x;
    const int wave = tid >> 5;
    const int lane = tid & 31;
    const int ml   = lane & 15;   // M (for A) / N (for B,C) within 16x16 tile
    const int kh   = lane >> 4;   // K-half selector

    const int m0 = blockIdx.y * TM + wave * 32;
    const int n0 = blockIdx.x * TN;

    // staging assignment: 16 bytes per thread per K-step
    const int scol = tid >> 2;          // 0..63
    const int sks  = (tid & 3) * 8;     // 0,8,16,24
    const _Float16* bsrc = &BT[(size_t)(n0 + scol) * K + sks];
    _Float16* sdst0 = &ldsB[0][scol * LDSB_STRIDE + sks];
    _Float16* sdst1 = &ldsB[1][scol * LDSB_STRIDE + sks];

    v8f acc[2][4] = {};

    // prologue: stage first B tile into buffer 0
    *(h8*)sdst0 = *(const h8*)bsrc;
    __syncthreads();

    int buf = 0;
    for (int k0 = 0; k0 < K; k0 += TK, buf ^= 1) {
        // stage next tile into the other buffer (overlaps with WMMA below)
        if (k0 + TK < K) {
            *(h8*)(buf ? sdst0 : sdst1) = *(const h8*)(bsrc + k0 + TK);
        }

        // A fragments (global, row-major; ISA 16-bit A lane layout)
        v16h a[2];
#pragma unroll
        for (int mt = 0; mt < 2; ++mt)
            a[mt] = load_a_frag(A + (size_t)(m0 + mt * 16 + ml) * K + k0, kh);

        // B fragments (LDS): preload all 4 so ds_loads issue together
        const _Float16* lb = &ldsB[buf][0];
        v16h b[4];
#pragma unroll
        for (int nt = 0; nt < 4; ++nt)
            b[nt] = load_b_frag(lb + (nt * 16 + ml) * LDSB_STRIDE + kh * 8);

#pragma unroll
        for (int mt = 0; mt < 2; ++mt)
#pragma unroll
            for (int nt = 0; nt < 4; ++nt)
                acc[mt][nt] = __builtin_amdgcn_wmma_f32_16x16x32_f16(
                    false, a[mt], false, b[nt], (short)0, acc[mt][nt], false, false);

        __syncthreads();
    }

    // --- epilogue: bias + relu, store f16 per C/D VGPR layout ---
#pragma unroll
    for (int mt = 0; mt < 2; ++mt) {
#pragma unroll
        for (int nt = 0; nt < 4; ++nt) {
            int col = n0 + nt * 16 + ml;
            float bv = bias[col];
#pragma unroll
            for (int r = 0; r < 8; ++r) {
                int m = m0 + mt * 16 + kh * 8 + r;
                float v = acc[mt][nt][r] + bv;
                v = v > 0.f ? v : 0.f;
                C[(size_t)m * N + col] = (_Float16)v;
            }
        }
    }
}

// ---------------------------------------------------------------------------
// Final layer: mlp_y = h3 . W4 + b4 ; out = sigmoid(bias + lin + 0.5*S + mlp_y)
// One wave per row.
// ---------------------------------------------------------------------------
__global__ __launch_bounds__(256) void final_layer(const _Float16* __restrict__ h3,
                                                   const float* __restrict__ W4,
                                                   const float* __restrict__ b4,
                                                   const float* __restrict__ bias0,
                                                   const float* __restrict__ lin,
                                                   const float* __restrict__ S,
                                                   float* __restrict__ out) {
    const int row  = blockIdx.x * 8 + (threadIdx.x >> 5);
    const int lane = threadIdx.x & 31;
    const _Float16* hp = h3 + (size_t)row * 512;
    float s = 0.f;
#pragma unroll
    for (int t = 0; t < 16; ++t) {
        int j = t * 32 + lane;
        s += (float)hp[j] * W4[j];
    }
#pragma unroll
    for (int off = 16; off >= 1; off >>= 1) s += __shfl_xor(s, off, 32);
    if (lane == 0) {
        float fm = bias0[0] + lin[row] + 0.5f * S[0];
        float z  = fm + s + b4[0];
        out[row] = 1.0f / (1.0f + expf(-z));
    }
}

// ---------------------------------------------------------------------------
extern "C" void kernel_launch(void* const* d_in, const int* in_sizes, int n_in,
                              void* d_out, int out_size, void* d_ws, size_t ws_size,
                              hipStream_t stream) {
    (void)in_sizes; (void)n_in; (void)out_size; (void)ws_size;
    // setup_inputs order: x, bias, fc, emb_table, W1,b1, W2,b2, W3,b3, W4,b4
    const float* x    = (const float*)d_in[0];
    const float* bias = (const float*)d_in[1];
    const float* fc   = (const float*)d_in[2];
    const float* emb  = (const float*)d_in[3];
    const float* W1   = (const float*)d_in[4];
    const float* b1   = (const float*)d_in[5];
    const float* W2   = (const float*)d_in[6];
    const float* b2   = (const float*)d_in[7];
    const float* W3   = (const float*)d_in[8];
    const float* b3   = (const float*)d_in[9];
    const float* W4   = (const float*)d_in[10];
    const float* b4   = (const float*)d_in[11];
    float* out = (float*)d_out;

    char* ws = (char*)d_ws;
    const size_t MB = 1024ull * 1024ull;
    float*    linArr  = (float*)(ws + 0);                 //  64 KB
    float*    rowterm = (float*)(ws + 64 * 1024);         //  64 KB
    float*    Ssum    = (float*)(ws + 128 * 1024);        //   4 B
    _Float16* W1T     = (_Float16*)(ws + 1 * MB);         //   2 MB (2048 x 512)
    _Float16* W2T     = (_Float16*)(ws + 3 * MB);         //   4 MB (1024 x 2048)
    _Float16* W3T     = (_Float16*)(ws + 7 * MB);         //   1 MB (512 x 1024)
    _Float16* embf16  = (_Float16*)(ws + 16 * MB);        //  16 MB (B x 512)
    _Float16* h1      = (_Float16*)(ws + 32 * MB);        //  64 MB (B x 2048)
    _Float16* h2      = (_Float16*)(ws + 96 * MB);        //  32 MB (B x 1024)
    _Float16* h3      = embf16;                           //  reuse (dead after L1)

    // 1) weight convert/transpose
    convert_wt<<<(512 * 2048 + 255) / 256, 256, 0, stream>>>(W1, W1T, 512, 2048);
    convert_wt<<<(2048 * 1024 + 255) / 256, 256, 0, stream>>>(W2, W2T, 2048, 1024);
    convert_wt<<<(1024 * 512 + 255) / 256, 256, 0, stream>>>(W3, W3T, 1024, 512);

    // 2) gather + FM row terms
    gather_fm<<<BATCH / 8, 256, 0, stream>>>(x, fc, emb, embf16, linArr, rowterm);

    // 3) batch-scalar FM interaction term (deterministic)
    reduce_sum<<<1, 256, 0, stream>>>(rowterm, Ssum, BATCH);

    // 4) MLP via WMMA
    gemm_relu_wmma<<<dim3(2048 / TN, BATCH / TM), 256, 0, stream>>>(
        embf16, W1T, b1, h1, BATCH, 2048, 512);
    gemm_relu_wmma<<<dim3(1024 / TN, BATCH / TM), 256, 0, stream>>>(
        h1, W2T, b2, h2, BATCH, 1024, 2048);
    gemm_relu_wmma<<<dim3(512 / TN, BATCH / TM), 256, 0, stream>>>(
        h2, W3T, b3, h3, BATCH, 512, 1024);

    // 5) final dot + FM + sigmoid
    final_layer<<<BATCH / 8, 256, 0, stream>>>(h3, W4, b4, bias, linArr, Ssum, out);
}
